// AugmentedLSTMCell_10565619549064
// MI455X (gfx1250) — compile-verified, matrix operand order checked by reading, not run on hardware
//
#include <hip/hip_runtime.h>
#include <hip/hip_bf16.h>

// ---------- types ----------
typedef __attribute__((ext_vector_type(16))) __bf16 v16bf;
typedef __attribute__((ext_vector_type(8)))  float  v8f;
typedef __attribute__((ext_vector_type(4)))  unsigned int u32x4;

// ---------- problem sizes ----------
#define BSZ   16384
#define EDIM  1024
#define HDIM  1024
#define KDIM  2048          // E + H after packing
#define NG    6             // i, f, m, o, hw, hw_proj
#define WROWS 6144          // 6*H rows of packed weight

// ---------- GEMM tiling ----------
#define BM    256           // rows per block (8 waves x 32 rows)
#define BN    16            // columns per gate per block
#define KT    64            // K per LDS stage (two wmma k-steps)
#define LPAD  8
#define LSTR  (KT + LPAD)   // 72 bf16 row stride in LDS
#define WM    32            // rows per wave (2 x 16 tiles)
#define NSTAGE (KDIM / KT)  // 32

// ---------- CDNA5 async global->LDS path (guarded) ----------
#if defined(__gfx1250__) && __has_builtin(__builtin_amdgcn_global_load_async_to_lds_b128)
#define USE_ASYNC 1
#else
#define USE_ASYNC 0
#endif

#if USE_ASYNC
// builtin signature (from hipcc diagnostic): param0 is
// 'int __attribute__((vector_size(16))) __device__ *'  -> AS(1) int4*
typedef int v4i_ __attribute__((vector_size(16)));
typedef __attribute__((address_space(1))) v4i_ gv4i;
typedef __attribute__((address_space(3))) v4i_ lv4i;
__device__ __forceinline__ void async_b128(const void* gsrc, void* ldst) {
  __builtin_amdgcn_global_load_async_to_lds_b128(
      (gv4i*)gsrc, (lv4i*)ldst, 0, 0);
}
__device__ __forceinline__ void wait_async0() {
#if __has_builtin(__builtin_amdgcn_s_wait_asynccnt)
  __builtin_amdgcn_s_wait_asynccnt(0);
#else
  asm volatile("s_wait_asynccnt 0x0" ::: "memory");
#endif
}
#else
__device__ __forceinline__ void wait_async0() {}
#endif

__device__ __forceinline__ float sigf(float x) {
  return 1.0f / (1.0f + __expf(-x));
}
__device__ __forceinline__ float tanh_fast(float x) {
  return 2.0f / (1.0f + __expf(-2.0f * x)) - 1.0f;
}

// ---------------- pack kernels ----------------
__global__ __launch_bounds__(256) void pack_a_kernel(
    const float* __restrict__ x, const float* __restrict__ h,
    __bf16* __restrict__ a) {
  long long idx = (long long)blockIdx.x * 256 + threadIdx.x;
  int b = (int)(idx >> 11);
  int k = (int)(idx & 2047);
  float v = (k < EDIM) ? x[(long long)b * EDIM + k]
                       : h[(long long)b * HDIM + (k - EDIM)];
  a[idx] = (__bf16)v;
}

__global__ __launch_bounds__(256) void pack_w_kernel(
    const float* __restrict__ Wi, const float* __restrict__ Ws,
    __bf16* __restrict__ w) {
  long long idx = (long long)blockIdx.x * 256 + threadIdx.x;
  int j = (int)(idx >> 11);
  int k = (int)(idx & 2047);
  float v;
  if (k < EDIM)          v = Wi[(long long)j * EDIM + k];
  else if (j < 5 * HDIM) v = Ws[(long long)j * HDIM + (k - EDIM)];
  else                   v = 0.0f;
  w[idx] = (__bf16)v;
}

__global__ __launch_bounds__(256) void pack_bias_kernel(
    const float* __restrict__ bi, const float* __restrict__ bs,
    float* __restrict__ bias) {
  int j = blockIdx.x * 256 + threadIdx.x;
  float v = bi[j];
  if (j < 5 * HDIM) v += bs[j];
  bias[j] = v;
}

// ---------------- fused WMMA GEMM + LSTM epilogue ----------------
// grid: (HDIM/BN, BSZ/BM) ; block: 256 threads (8 waves)
__global__ __launch_bounds__(256, 1) void lstm_wmma_kernel(
    const __bf16* __restrict__ A,    // [BSZ, KDIM]
    const __bf16* __restrict__ W,    // [WROWS, KDIM]
    const float*  __restrict__ bias, // [WROWS]
    const float*  __restrict__ c,    // [BSZ, HDIM]
    float* __restrict__ out,
    float* __restrict__ mem) {
  __shared__ __align__(16) __bf16 ldsA[2][BM * LSTR];       // 2*256*72*2B = 72 KB
  __shared__ __align__(16) __bf16 ldsB[2][NG * BN * LSTR];  // 2* 96*72*2B = 27 KB

  const int tid  = threadIdx.x;
  const int lane = tid & 31;
  const int wave = tid >> 5;
  const int rowBase = blockIdx.y * BM;
  const int colBase = blockIdx.x * BN;

#if !USE_ASYNC
  u32x4 pa[8];  // fallback register prefetch: A = 2048 chunks / 256 thr
  u32x4 pb[3];  //                             B =  768 chunks / 256 thr
#endif

  // Copy one K-stage (KT=64 columns) of A and B into LDS buffer `buf`.
  // A: 256 rows x 8 b128-chunks; B: 96 rows x 8 chunks. Perfectly uniform.
  auto copyStage = [&](int buf, int ks) {
#if USE_ASYNC
#pragma unroll
    for (int i = 0; i < 8; ++i) {
      int e = i * 256 + tid;
      int r = e >> 3, ch = e & 7;
      async_b128(A + (long long)(rowBase + r) * KDIM + ks * KT + ch * 8,
                 &ldsA[buf][r * LSTR + ch * 8]);
    }
#pragma unroll
    for (int i = 0; i < 3; ++i) {
      int e = i * 256 + tid;
      int r = e >> 3, ch = e & 7;
      int g = r >> 4, n = r & 15;
      async_b128(W + (long long)(g * HDIM + colBase + n) * KDIM + ks * KT + ch * 8,
                 &ldsB[buf][r * LSTR + ch * 8]);
    }
#else
#pragma unroll
    for (int i = 0; i < 8; ++i) {
      int e = i * 256 + tid;
      int r = e >> 3, ch = e & 7;
      pa[i] = *(const u32x4*)(A + (long long)(rowBase + r) * KDIM + ks * KT + ch * 8);
    }
#pragma unroll
    for (int i = 0; i < 3; ++i) {
      int e = i * 256 + tid;
      int r = e >> 3, ch = e & 7;
      int g = r >> 4, n = r & 15;
      pb[i] = *(const u32x4*)(W + (long long)(g * HDIM + colBase + n) * KDIM + ks * KT + ch * 8);
    }
#pragma unroll
    for (int i = 0; i < 8; ++i) {
      int e = i * 256 + tid;
      int r = e >> 3, ch = e & 7;
      *(u32x4*)&ldsA[buf][r * LSTR + ch * 8] = pa[i];
    }
#pragma unroll
    for (int i = 0; i < 3; ++i) {
      int e = i * 256 + tid;
      int r = e >> 3, ch = e & 7;
      *(u32x4*)&ldsB[buf][r * LSTR + ch * 8] = pb[i];
    }
#endif
  };

  v8f acc[2][NG];
#pragma unroll
  for (int t = 0; t < 2; ++t)
#pragma unroll
    for (int g = 0; g < NG; ++g) acc[t][g] = (v8f)(0.0f);

  // ISA fragment layout for 16-bit A (16x32) / B (32x16):
  // lane<16 : row/col = lane,     K = {kb..kb+7, kb+16..kb+23}, kb=0
  // lane>=16: row/col = lane-16,  kb=8
  const int fr = lane & 15;
  const int kb = (lane >> 4) << 3;

  union Frag { u32x4 q[2]; v16bf v; };

  copyStage(0, 0);

  for (int ks = 0; ks < NSTAGE; ++ks) {
    const int buf = ks & 1;
    wait_async0();        // this wave's copies into `buf` (issued last iter) done
    __syncthreads();      // everyone's copies done; prev reads of buf^1 done
    if (ks + 1 < NSTAGE) copyStage(buf ^ 1, ks + 1);  // overlap with WMMAs

#pragma unroll
    for (int ksub = 0; ksub < 2; ++ksub) {
      const int ko = ksub * 32;
      Frag afr[2];
#pragma unroll
      for (int t = 0; t < 2; ++t) {
        const __bf16* base = &ldsA[buf][(wave * WM + t * 16 + fr) * LSTR + ko + kb];
        afr[t].q[0] = *(const u32x4*)(base);
        afr[t].q[1] = *(const u32x4*)(base + 16);
      }
      Frag bfr[NG];   // batch all 6 gate fragments -> one dscnt wait, wmma b2b
#pragma unroll
      for (int g = 0; g < NG; ++g) {
        const __bf16* base = &ldsB[buf][(g * BN + fr) * LSTR + ko + kb];
        bfr[g].q[0] = *(const u32x4*)(base);
        bfr[g].q[1] = *(const u32x4*)(base + 16);
      }
#pragma unroll
      for (int g = 0; g < NG; ++g)
#pragma unroll
        for (int t = 0; t < 2; ++t)
          acc[t][g] = __builtin_amdgcn_wmma_f32_16x16x32_bf16(
              false, afr[t].v, false, bfr[g].v, (short)0, acc[t][g], false, false);
    }
  }

  // ---------------- epilogue: lane-local LSTM math ----------------
  const int n   = lane & 15;
  const int col = colBase + n;
  float bv[NG];
#pragma unroll
  for (int g = 0; g < NG; ++g) bv[g] = bias[g * HDIM + col];

  const int mo = (lane >> 4) << 3;  // C/D layout: M = v + (lane>=16 ? 8 : 0)
#pragma unroll
  for (int t = 0; t < 2; ++t) {
#pragma unroll
    for (int v = 0; v < 8; ++v) {
      const int row = rowBase + wave * WM + t * 16 + mo + v;
      const long long o = (long long)row * HDIM + col;
      float ig = sigf(acc[t][0][v] + bv[0]);
      float fg = sigf(acc[t][1][v] + bv[1]);
      float mi = tanh_fast(acc[t][2][v] + bv[2]);
      float og = sigf(acc[t][3][v] + bv[3]);
      float hw = sigf(acc[t][4][v] + bv[4]);
      float hp = acc[t][5][v] + bv[5];
      float cv = c[o];
      float me = ig * mi + fg * cv;
      float ov = og * tanh_fast(me);
      out[o] = hw * ov + (1.0f - hw) * hp;
      mem[o] = me;
    }
  }
}

// ---------------- launcher ----------------
extern "C" void kernel_launch(void* const* d_in, const int* in_sizes, int n_in,
                              void* d_out, int out_size, void* d_ws, size_t ws_size,
                              hipStream_t stream) {
  const float* x  = (const float*)d_in[0];
  const float* h  = (const float*)d_in[1];
  const float* c  = (const float*)d_in[2];
  const float* Wi = (const float*)d_in[3];
  const float* bi = (const float*)d_in[4];
  const float* Ws = (const float*)d_in[5];
  const float* bs = (const float*)d_in[6];

  char* ws = (char*)d_ws;
  __bf16* Abf   = (__bf16*)ws;                                   // 64 MB
  __bf16* Wbf   = (__bf16*)(ws + (size_t)BSZ * KDIM * 2);        // 24 MB
  float*  biasf = (float*)(ws + (size_t)BSZ * KDIM * 2
                              + (size_t)WROWS * KDIM * 2);       // 24 KB

  float* out = (float*)d_out;
  float* mem = out + (size_t)BSZ * HDIM;

  pack_a_kernel<<<(BSZ * (long long)KDIM) / 256, 256, 0, stream>>>(x, h, Abf);
  pack_w_kernel<<<((long long)WROWS * KDIM) / 256, 256, 0, stream>>>(Wi, Ws, Wbf);
  pack_bias_kernel<<<WROWS / 256, 256, 0, stream>>>(bi, bs, biasf);

  dim3 grid(HDIM / BN, BSZ / BM);  // (64, 64)
  lstm_wmma_kernel<<<grid, 256, 0, stream>>>(Abf, Wbf, biasf, c, out, mem);
}
